// FastWeightAttention_773094113980
// MI455X (gfx1250) — matrix operand order repaired and checked by reading, not compiled, not run on hardware
//
#include <hip/hip_runtime.h>
#include <hip/hip_fp16.h>

#define DEVFN static __device__ __forceinline__

typedef __attribute__((ext_vector_type(16))) _Float16 v16h;
typedef __attribute__((ext_vector_type(8)))  _Float16 v8h;
typedef __attribute__((ext_vector_type(8)))  float    v8f;

constexpr int B = 2, N = 2048, D = 512, H = 8, DH = 64;
constexpr float ATT_SCALE = 0.125f;   // 64^-0.5
constexpr float MAXLR = 0.01f;
#define NEG_INF (-__builtin_inff())

DEVFN int lane_id() { return (int)(threadIdx.x & 31u); }
DEVFN v8f vzero() { v8f z = {0.f,0.f,0.f,0.f,0.f,0.f,0.f,0.f}; return z; }

DEVFN v8f wmma16(v16h a, v16h b, v8f c) {
  return __builtin_amdgcn_wmma_f32_16x16x32_f16(false, a, false, b, (short)0, c, false, false);
}

// A fragment (16x32 f16, row-major source, ld in elements).
// lane L: m = m0 + (L&15); elems 0..7 -> K = kb..kb+7, elems 8..15 -> K = kb+16..kb+23,
// kb = k0 + (L>=16 ? 8 : 0).
DEVFN v16h frag_a(const _Float16* A, int lda, int m0, int k0) {
  int l = lane_id();
  const _Float16* p = A + (size_t)(m0 + (l & 15)) * lda + (size_t)(k0 + ((l >> 4) << 3));
  v8h lo = *(const v8h*)p;
  v8h hi = *(const v8h*)(p + 16);
  v16h r;
#pragma unroll
  for (int i = 0; i < 8; ++i) { r[i] = lo[i]; r[i + 8] = hi[i]; }
  return r;
}

// B fragment (32x16 f16) loaded from Bt where Bt[n][k] = B[k][n] (k contiguous).
// lane L: n = n0 + (L&15); elems 0..7 -> K = kb..kb+7, elems 8..15 -> K = kb+16..kb+23,
// kb = k0 + (L>=16 ? 8 : 0).
DEVFN v16h frag_b(const _Float16* Bt, int ldb, int n0, int k0) {
  int l = lane_id();
  const _Float16* p = Bt + (size_t)(n0 + (l & 15)) * ldb + (size_t)(k0 + ((l >> 4) << 3));
  v8h lo = *(const v8h*)p;
  v8h hi = *(const v8h*)(p + 16);
  v16h r;
#pragma unroll
  for (int i = 0; i < 8; ++i) { r[i] = lo[i]; r[i + 8] = hi[i]; }
  return r;
}

// D fragment store (16x16 f32): lane L: col = n0+(L&15); rows m0 + (L>=16?8:0) + i
DEVFN void store_d(float* C, int ldc, int m0, int n0, v8f d) {
  int l = lane_id();
  int n = n0 + (l & 15);
  int mb = m0 + ((l >> 4) << 3);
#pragma unroll
  for (int i = 0; i < 8; ++i) C[(size_t)(mb + i) * ldc + n] = d[i];
}

// ---------------------------------------------------------------------------
// Weight prep: f32 -> f16, with transposed copies so WMMA B-operands are contiguous.
__global__ __launch_bounds__(256) void k_prep(
    const float* wq, const float* wk, const float* wv, const float* wo, const float* tvw,
    _Float16* wqT, _Float16* wkT, _Float16* wvT,
    _Float16* wo16, _Float16* woT16, _Float16* tvw16) {
  int tid = blockIdx.x * 256 + threadIdx.x;
  const int NW = H * DH * D;                   // 262144 per projection weight
  if (tid < 3 * NW) {
    int w = tid / NW, r = tid % NW;
    int h = r / (DH * D), r2 = r % (DH * D);
    int e = r2 / D, d = r2 % D;
    const float* src = (w == 0) ? wq : (w == 1) ? wk : wv;
    _Float16* dst = (w == 0) ? wqT : (w == 1) ? wkT : wvT;
    dst[((size_t)h * DH + e) * D + d] = (_Float16)src[((size_t)h * D + d) * DH + e];
  } else if (tid < 3 * NW + D * D) {
    int r = tid - 3 * NW;                      // wo flat [(h,e)][d]
    float v = wo[r];
    wo16[r] = (_Float16)v;
    int k = r / D, d = r % D;
    woT16[(size_t)d * D + k] = (_Float16)v;    // woT16[d][(h,e)]
  } else if (tid < 3 * NW + 2 * D * D) {
    int r = tid - 3 * NW - D * D;
    tvw16[r] = (_Float16)tvw[r];               // tv_w[o][d] is already Bt for tv GEMM
  }
}

// ---------------------------------------------------------------------------
// RMSNorm: t32, t16 (row-major) and tT16 (d-major over flat b*n)
__global__ __launch_bounds__(256) void k_rmsnorm(
    const float* tokens, const float* rmsw, float* t32, _Float16* t16, _Float16* tT16) {
  int row = blockIdx.x;                        // b*N + i
  const float* x = tokens + (size_t)row * D;
  __shared__ float red[256];
  float s = 0.f;
  for (int d = threadIdx.x; d < D; d += 256) { float v = x[d]; s += v * v; }
  red[threadIdx.x] = s; __syncthreads();
  for (int off = 128; off; off >>= 1) {
    if ((int)threadIdx.x < off) red[threadIdx.x] += red[threadIdx.x + off];
    __syncthreads();
  }
  float r = rsqrtf(red[0] / (float)D + 1.1920929e-07f);
  for (int d = threadIdx.x; d < D; d += 256) {
    float v = x[d] * r * rmsw[d];
    t32[(size_t)row * D + d] = v;
    t16[(size_t)row * D + d] = (_Float16)v;
    tT16[(size_t)d * (B * N) + row] = (_Float16)v;
  }
}

// ---------------------------------------------------------------------------
// QKV projections: per (b,h), 64-row tile per block; also writes transposed copies.
DEVFN void store_rt(_Float16* rowm, _Float16* tr, int m0, int n0, v8f d) {
  int l = lane_id();
  int n = n0 + (l & 15);
  int mb = m0 + ((l >> 4) << 3);
#pragma unroll
  for (int i = 0; i < 8; ++i) {
    _Float16 h = (_Float16)d[i];
    rowm[(size_t)(mb + i) * DH + n] = h;       // [n][e]
    tr[(size_t)n * N + (mb + i)] = h;          // [e][n]
  }
}

__global__ __launch_bounds__(256) void k_qkv(
    const _Float16* t16, const _Float16* wqT, const _Float16* wkT, const _Float16* wvT,
    _Float16* q16, _Float16* k16, _Float16* v16,
    _Float16* qT16, _Float16* kT16, _Float16* vT16) {
  int b = blockIdx.z, h = blockIdx.y;
  int w = (int)(threadIdx.x >> 5);
  int m0 = blockIdx.x * 64 + (w >> 1) * 16;
  int n0 = (w & 1) * 32;
  const _Float16* A  = t16 + (size_t)b * N * D;
  const _Float16* Bq = wqT + (size_t)h * DH * D;
  const _Float16* Bk = wkT + (size_t)h * DH * D;
  const _Float16* Bv = wvT + (size_t)h * DH * D;
  v8f aq0 = vzero(), aq1 = vzero(), ak0 = vzero(), ak1 = vzero(), av0 = vzero(), av1 = vzero();
  for (int k = 0; k < D; k += 32) {
    v16h a = frag_a(A, D, m0, k);
    aq0 = wmma16(a, frag_b(Bq, D, n0, k), aq0);
    aq1 = wmma16(a, frag_b(Bq, D, n0 + 16, k), aq1);
    ak0 = wmma16(a, frag_b(Bk, D, n0, k), ak0);
    ak1 = wmma16(a, frag_b(Bk, D, n0 + 16, k), ak1);
    av0 = wmma16(a, frag_b(Bv, D, n0, k), av0);
    av1 = wmma16(a, frag_b(Bv, D, n0 + 16, k), av1);
  }
  size_t bh = (size_t)b * H + h;
  store_rt(q16 + bh * N * DH, qT16 + bh * DH * N, m0, n0, aq0);
  store_rt(q16 + bh * N * DH, qT16 + bh * DH * N, m0, n0 + 16, aq1);
  store_rt(k16 + bh * N * DH, kT16 + bh * DH * N, m0, n0, ak0);
  store_rt(k16 + bh * N * DH, kT16 + bh * DH * N, m0, n0 + 16, ak1);
  store_rt(v16 + bh * N * DH, vT16 + bh * DH * N, m0, n0, av0);
  store_rt(v16 + bh * N * DH, vT16 + bh * DH * N, m0, n0 + 16, av1);
}

// ---------------------------------------------------------------------------
// Flash attention forward. Block = 4 waves x 16 rows. Stores out (2 layouts) + LSE.
__global__ __launch_bounds__(128) void k_attn_fwd(
    const _Float16* q16, const _Float16* k16, const _Float16* vT16,
    _Float16* out16, _Float16* outT16, float* lse) {
  int b = blockIdx.z, h = blockIdx.y, bx = blockIdx.x;
  int w = (int)(threadIdx.x >> 5), l = lane_id();
  size_t bh = (size_t)b * H + h;
  const _Float16* Q  = q16 + bh * N * DH;
  const _Float16* K  = k16 + bh * N * DH;
  const _Float16* VT = vT16 + bh * DH * N;
  int i0 = bx * 64 + w * 16;
  int mb = (l >> 4) << 3;
  __shared__ __align__(16) _Float16 sP[4][16 * 64];
  _Float16* Pt = sP[w];
  v8f O[4]; float mrow[8], lrow[8];
#pragma unroll
  for (int i = 0; i < 4; ++i) O[i] = vzero();
#pragma unroll
  for (int i = 0; i < 8; ++i) { mrow[i] = NEG_INF; lrow[i] = 0.f; }

  v16h a0 = frag_a(Q, DH, i0, 0), a1 = frag_a(Q, DH, i0, 32);
  for (int jt = 0; jt <= bx; ++jt) {
    int j0 = jt * 64;
    v8f S[4];
#pragma unroll
    for (int nj = 0; nj < 4; ++nj) {
      v8f s = vzero();
      s = wmma16(a0, frag_b(K, DH, j0 + nj * 16, 0), s);
      s = wmma16(a1, frag_b(K, DH, j0 + nj * 16, 32), s);
      S[nj] = s;
    }
    float mnew[8], sc[8];
#pragma unroll
    for (int i = 0; i < 8; ++i) {
      int gi = i0 + mb + i;
      float mx = NEG_INF;
#pragma unroll
      for (int nj = 0; nj < 4; ++nj) {
        int gj = j0 + nj * 16 + (l & 15);
        float s = S[nj][i] * ATT_SCALE;
        if (gj > gi) s = NEG_INF;              // causal mask
        S[nj][i] = s;
        mx = fmaxf(mx, s);
      }
      for (int off = 8; off >= 1; off >>= 1) mx = fmaxf(mx, __shfl_xor(mx, off));
      mnew[i] = fmaxf(mrow[i], mx);
      sc[i] = __expf(mrow[i] - mnew[i]);
    }
#pragma unroll
    for (int i = 0; i < 8; ++i) {
      float rs = 0.f;
#pragma unroll
      for (int nj = 0; nj < 4; ++nj) {
        float p = __expf(S[nj][i] - mnew[i]);
        rs += p;
        Pt[(mb + i) * 64 + nj * 16 + (l & 15)] = (_Float16)p;
      }
      for (int off = 8; off >= 1; off >>= 1) rs += __shfl_xor(rs, off);
      lrow[i] = lrow[i] * sc[i] + rs;
      mrow[i] = mnew[i];
#pragma unroll
      for (int ne = 0; ne < 4; ++ne) O[ne][i] *= sc[i];
    }
    v16h p0 = frag_a(Pt, 64, 0, 0), p1 = frag_a(Pt, 64, 0, 32);
#pragma unroll
    for (int ne = 0; ne < 4; ++ne) {
      O[ne] = wmma16(p0, frag_b(VT, N, ne * 16, j0), O[ne]);
      O[ne] = wmma16(p1, frag_b(VT, N, ne * 16, j0 + 32), O[ne]);
    }
  }
#pragma unroll
  for (int i = 0; i < 8; ++i) {
    float inv = 1.f / lrow[i];
    int gi = i0 + mb + i;
#pragma unroll
    for (int ne = 0; ne < 4; ++ne) {
      float o = O[ne][i] * inv;
      int e = ne * 16 + (l & 15);
      out16[((size_t)b * N + gi) * (H * DH) + h * DH + e] = (_Float16)o;
      outT16[bh * DH * N + (size_t)e * N + gi] = (_Float16)o;
    }
    if ((l & 15) == 0) lse[bh * N + gi] = mrow[i] + logf(lrow[i]);
  }
}

// ---------------------------------------------------------------------------
// pred = out16 (4096x512) @ woT16 (Bt 512x512), f32 out -> d_out
__global__ __launch_bounds__(256) void k_gemm_pred(
    const _Float16* out16, const _Float16* woT16, float* pred) {
  int wid = blockIdx.x * 8 + (int)(threadIdx.x >> 5);   // 8192 tiles
  int m0 = (wid >> 5) * 16, n0 = (wid & 31) * 16;
  v8f acc = vzero();
  for (int k = 0; k < D; k += 32)
    acc = wmma16(frag_a(out16, D, m0, k), frag_b(woT16, D, n0, k), acc);
  store_d(pred, D, m0, n0, acc);
}

// tv = t[:,1:] @ tv_w^T (rows shifted by one within each batch; last row clamped/ignored)
DEVFN v16h frag_a_shift1(const _Float16* A, int lda, int m0, int k0) {
  int l = lane_id();
  int r = m0 + (l & 15);
  int b = r >> 11, i = r & 2047;
  int i2 = (i + 1 < N) ? (i + 1) : (N - 1);
  const _Float16* p = A + (size_t)((b << 11) + i2) * lda + (size_t)(k0 + ((l >> 4) << 3));
  v8h lo = *(const v8h*)p;
  v8h hi = *(const v8h*)(p + 16);
  v16h rr;
#pragma unroll
  for (int ii = 0; ii < 8; ++ii) { rr[ii] = lo[ii]; rr[ii + 8] = hi[ii]; }
  return rr;
}

__global__ __launch_bounds__(256) void k_gemm_tv(
    const _Float16* t16, const _Float16* tvw16, float* tv) {
  int wid = blockIdx.x * 8 + (int)(threadIdx.x >> 5);
  int m0 = (wid >> 5) * 16, n0 = (wid & 31) * 16;
  v8f acc = vzero();
  for (int k = 0; k < D; k += 32)
    acc = wmma16(frag_a_shift1(t16, D, m0, k), frag_b(tvw16, D, n0, k), acc);
  store_d(tv, D, m0, n0, acc);
}

// ---------------------------------------------------------------------------
// LayerNorm(tv) -> target; lr = sigmoid(t . lr_w)*MAXLR; error = (target-pred)*lr
// Row i = N-1 is zero-filled (truncation in reference).
__global__ __launch_bounds__(256) void k_error(
    const float* tv, const float* pred, const float* t32, const float* lrw,
    _Float16* err16, _Float16* errT16) {
  int row = blockIdx.x;                        // b*N + i
  int i = row & (N - 1);
  if (i == N - 1) {
    for (int d = threadIdx.x; d < D; d += 256) {
      err16[(size_t)row * D + d] = (_Float16)0.f;
      errT16[(size_t)d * (B * N) + row] = (_Float16)0.f;
    }
    return;
  }
  __shared__ float r1[256], r2[256], r3[256];
  const float* x = tv + (size_t)row * D;
  float s1 = 0.f, s2 = 0.f, s3 = 0.f;
  for (int d = threadIdx.x; d < D; d += 256) {
    float v = x[d];
    s1 += v; s2 += v * v;
    s3 += t32[(size_t)row * D + d] * lrw[d];
  }
  r1[threadIdx.x] = s1; r2[threadIdx.x] = s2; r3[threadIdx.x] = s3;
  __syncthreads();
  for (int off = 128; off; off >>= 1) {
    if ((int)threadIdx.x < off) {
      r1[threadIdx.x] += r1[threadIdx.x + off];
      r2[threadIdx.x] += r2[threadIdx.x + off];
      r3[threadIdx.x] += r3[threadIdx.x + off];
    }
    __syncthreads();
  }
  float mu = r1[0] / (float)D;
  float var = r2[0] / (float)D - mu * mu;
  float rstd = rsqrtf(var + 1e-5f);
  float lr = (1.f / (1.f + __expf(-r3[0]))) * MAXLR;
  for (int d = threadIdx.x; d < D; d += 256) {
    float e = ((x[d] - mu) * rstd - pred[(size_t)row * D + d]) * lr;
    err16[(size_t)row * D + d] = (_Float16)e;
    errT16[(size_t)d * (B * N) + row] = (_Float16)e;
  }
}

// dout[b,h,i,e] = error[b,i,:] @ wo[(h,e),:] ; scatter-stores both layouts
__global__ __launch_bounds__(256) void k_gemm_dout(
    const _Float16* err16, const _Float16* wo16, _Float16* dout16, _Float16* doutT16) {
  int wid = blockIdx.x * 8 + (int)(threadIdx.x >> 5);
  int m0 = (wid >> 5) * 16, n0 = (wid & 31) * 16;
  v8f acc = vzero();
  for (int k = 0; k < D; k += 32)
    acc = wmma16(frag_a(err16, D, m0, k), frag_b(wo16, D, n0, k), acc);
  int l = lane_id();
  int n = n0 + (l & 15), mb = m0 + ((l >> 4) << 3);
  int h = n >> 6, e = n & 63;
#pragma unroll
  for (int i = 0; i < 8; ++i) {
    int r = mb + i, b = r >> 11, gi = r & (N - 1);
    _Float16 v = (_Float16)acc[i];
    size_t bh = (size_t)b * H + h;
    dout16[bh * N * DH + (size_t)gi * DH + e] = v;
    doutT16[bh * DH * N + (size_t)e * N + gi] = v;
  }
}

// delta[b,h,i] = sum_e dout*out ; one wave per row
__global__ __launch_bounds__(256) void k_delta(
    const _Float16* dout16, const _Float16* out16, float* delta) {
  int wid = blockIdx.x * 8 + (int)(threadIdx.x >> 5);  // 0..32767
  int l = lane_id();
  int bh = wid >> 11, gi = wid & (N - 1);
  int b = bh >> 3, h = bh & 7;
  float s = 0.f;
  for (int e = l; e < DH; e += 32)
    s += (float)dout16[(size_t)bh * N * DH + (size_t)gi * DH + e] *
         (float)out16[((size_t)b * N + gi) * (H * DH) + h * DH + e];
  for (int off = 16; off >= 1; off >>= 1) s += __shfl_xor(s, off);
  if (l == 0) delta[(size_t)bh * N + gi] = s;
}

// ---------------------------------------------------------------------------
// Backward (dq): block per (b,h,i-tile). Recompute P from LSE; dS via LDS re-layout.
__global__ __launch_bounds__(128) void k_bwd_dq(
    const _Float16* q16, const _Float16* k16, const _Float16* v16, const _Float16* kT16,
    const _Float16* dout16, const float* lse, const float* delta, _Float16* dqT16) {
  int b = blockIdx.z, h = blockIdx.y, bx = blockIdx.x;
  int w = (int)(threadIdx.x >> 5), l = lane_id();
  size_t bh = (size_t)b * H + h;
  const _Float16* Q  = q16 + bh * N * DH;
  const _Float16* K  = k16 + bh * N * DH;
  const _Float16* V  = v16 + bh * N * DH;
  const _Float16* KT = kT16 + bh * DH * N;
  const _Float16* DO = dout16 + bh * N * DH;
  int i0 = bx * 64 + w * 16;
  int mb = (l >> 4) << 3;
  float lse8[8], dl8[8];
#pragma unroll
  for (int i = 0; i < 8; ++i) {
    lse8[i] = lse[bh * N + i0 + mb + i];
    dl8[i] = delta[bh * N + i0 + mb + i];
  }
  __shared__ __align__(16) _Float16 sT[4][16 * 64];
  _Float16* St = sT[w];
  v8f dq[4];
#pragma unroll
  for (int i = 0; i < 4; ++i) dq[i] = vzero();
  v16h a0 = frag_a(Q, DH, i0, 0), a1 = frag_a(Q, DH, i0, 32);
  v16h d0 = frag_a(DO, DH, i0, 0), d1 = frag_a(DO, DH, i0, 32);
  for (int jt = 0; jt <= bx; ++jt) {
    int j0 = jt * 64;
#pragma unroll
    for (int nj = 0; nj < 4; ++nj) {
      v8f S = vzero(), dP = vzero();
      S = wmma16(a0, frag_b(K, DH, j0 + nj * 16, 0), S);
      S = wmma16(a1, frag_b(K, DH, j0 + nj * 16, 32), S);
      dP = wmma16(d0, frag_b(V, DH, j0 + nj * 16, 0), dP);
      dP = wmma16(d1, frag_b(V, DH, j0 + nj * 16, 32), dP);
      int gj = j0 + nj * 16 + (l & 15);
#pragma unroll
      for (int i = 0; i < 8; ++i) {
        int gi = i0 + mb + i;
        float p = (gj <= gi && gj != N - 1) ? __expf(S[i] * ATT_SCALE - lse8[i]) : 0.f;
        float ds = ATT_SCALE * p * (dP[i] - dl8[i]);
        St[(mb + i) * 64 + nj * 16 + (l & 15)] = (_Float16)ds;
      }
    }
    v16h s0 = frag_a(St, 64, 0, 0), s1 = frag_a(St, 64, 0, 32);
#pragma unroll
    for (int ne = 0; ne < 4; ++ne) {
      dq[ne] = wmma16(s0, frag_b(KT, N, ne * 16, j0), dq[ne]);
      dq[ne] = wmma16(s1, frag_b(KT, N, ne * 16, j0 + 32), dq[ne]);
    }
  }
#pragma unroll
  for (int ne = 0; ne < 4; ++ne) {
    int e = ne * 16 + (l & 15);
#pragma unroll
    for (int i = 0; i < 8; ++i)
      dqT16[bh * DH * N + (size_t)e * N + (i0 + mb + i)] = (_Float16)dq[ne][i];
  }
}

// Backward (dk, dv): block per (b,h,j-tile), transposed-P formulation.
__global__ __launch_bounds__(128) void k_bwd_dkv(
    const _Float16* q16, const _Float16* k16, const _Float16* v16, const _Float16* qT16,
    const _Float16* dout16, const _Float16* doutT16,
    const float* lse, const float* delta, _Float16* dkT16, _Float16* dvT16) {
  int b = blockIdx.z, h = blockIdx.y, bx = blockIdx.x;
  int w = (int)(threadIdx.x >> 5), l = lane_id();
  size_t bh = (size_t)b * H + h;
  const _Float16* Q   = q16 + bh * N * DH;
  const _Float16* K   = k16 + bh * N * DH;
  const _Float16* V   = v16 + bh * N * DH;
  const _Float16* QT  = qT16 + bh * DH * N;
  const _Float16* DO  = dout16 + bh * N * DH;
  const _Float16* DOT = doutT16 + bh * DH * N;
  int j0 = bx * 64 + w * 16;
  int mb = (l >> 4) << 3;
  __shared__ __align__(16) _Float16 sT[4][16 * 64];
  _Float16* Tt = sT[w];
  v8f dv[4], dk[4];
#pragma unroll
  for (int i = 0; i < 4; ++i) { dv[i] = vzero(); dk[i] = vzero(); }
  v16h ka0 = frag_a(K, DH, j0, 0), ka1 = frag_a(K, DH, j0, 32);
  v16h va0 = frag_a(V, DH, j0, 0), va1 = frag_a(V, DH, j0, 32);
  for (int it = bx; it < N / 64; ++it) {
    int i0 = it * 64;
    v8f Pf[4], Df[4];
#pragma unroll
    for (int ni = 0; ni < 4; ++ni) {
      v8f st = vzero(), dpt = vzero();
      st  = wmma16(ka0, frag_b(Q, DH, i0 + ni * 16, 0), st);
      st  = wmma16(ka1, frag_b(Q, DH, i0 + ni * 16, 32), st);
      dpt = wmma16(va0, frag_b(DO, DH, i0 + ni * 16, 0), dpt);
      dpt = wmma16(va1, frag_b(DO, DH, i0 + ni * 16, 32), dpt);
      int gi = i0 + ni * 16 + (l & 15);
      float lsec = lse[bh * N + gi];
      float dlc  = delta[bh * N + gi];
#pragma unroll
      for (int i = 0; i < 8; ++i) {
        int gj = j0 + mb + i;
        float p = (gj <= gi && gj != N - 1) ? __expf(st[i] * ATT_SCALE - lsec) : 0.f;
        Tt[(mb + i) * 64 + ni * 16 + (l & 15)] = (_Float16)p;   // P^T tile
        Pf[ni][i] = p;
        Df[ni][i] = dpt[i] - dlc;
      }
    }
    v16h p0 = frag_a(Tt, 64, 0, 0), p1 = frag_a(Tt, 64, 0, 32);
#pragma unroll
    for (int ne = 0; ne < 4; ++ne) {
      dv[ne] = wmma16(p0, frag_b(DOT, N, ne * 16, i0), dv[ne]);
      dv[ne] = wmma16(p1, frag_b(DOT, N, ne * 16, i0 + 32), dv[ne]);
    }
#pragma unroll
    for (int ni = 0; ni < 4; ++ni)
#pragma unroll
      for (int i = 0; i < 8; ++i)
        Tt[(mb + i) * 64 + ni * 16 + (l & 15)] =
            (_Float16)(ATT_SCALE * Pf[ni][i] * Df[ni][i]);       // dS^T tile
    v16h s0 = frag_a(Tt, 64, 0, 0), s1 = frag_a(Tt, 64, 0, 32);
#pragma unroll
    for (int ne = 0; ne < 4; ++ne) {
      dk[ne] = wmma16(s0, frag_b(QT, N, ne * 16, i0), dk[ne]);
      dk[ne] = wmma16(s1, frag_b(QT, N, ne * 16, i0 + 32), dk[ne]);
    }
  }
#pragma unroll
  for (int ne = 0; ne < 4; ++ne) {
    int e = ne * 16 + (l & 15);
#pragma unroll
    for (int i = 0; i < 8; ++i) {
      dvT16[bh * DH * N + (size_t)e * N + (j0 + mb + i)] = (_Float16)dv[ne][i];
      dkT16[bh * DH * N + (size_t)e * N + (j0 + mb + i)] = (_Float16)dk[ne][i];
    }
  }
}

// ---------------------------------------------------------------------------
// Weight-gradient GEMMs: dwq/dwk -> d_out, raw dwv/dwo -> ws (pre Newton-Schulz)
__global__ __launch_bounds__(256) void k_dw(
    const _Float16* tT16,
    const _Float16* dqT16, const _Float16* dkT16, const _Float16* dvT16,
    const _Float16* outT16, const _Float16* errT16,
    float* dwq_o, float* dwk_o, float* rawdwv, float* rawdwo) {
  int wid = blockIdx.x * 8 + (int)(threadIdx.x >> 5);   // 0..4095
  int which = wid >> 10;
  int r = wid & 1023;
  int h = r >> 7, rr = r & 127;
  v8f acc = vzero();
  if (which < 3) {
    const _Float16* G = (which == 0) ? dqT16 : (which == 1) ? dkT16 : dvT16;
    int m0 = (rr >> 2) * 16, n0 = (rr & 3) * 16;        // M=512(d), N=64(e)
    for (int b = 0; b < B; ++b) {
      const _Float16* Bt = G + ((size_t)b * H + h) * DH * N;
      for (int k = 0; k < N; k += 32)
        acc = wmma16(frag_a(tT16, B * N, m0, b * N + k), frag_b(Bt, N, n0, k), acc);
    }
    float* C = (which == 0) ? dwq_o : (which == 1) ? dwk_o : rawdwv;
    store_d(C + (size_t)h * D * DH, DH, m0, n0, acc);
  } else {
    int m0 = (rr >> 5) * 16, n0 = (rr & 31) * 16;       // M=64(e), N=512(d)
    for (int b = 0; b < B; ++b) {
      const _Float16* A = outT16 + ((size_t)b * H + h) * DH * N;
      for (int k = 0; k < N; k += 32)
        acc = wmma16(frag_a(A, N, m0, k), frag_b(errT16, B * N, n0, b * N + k), acc);
    }
    store_d(rawdwo + (size_t)h * DH * D, D, m0, n0, acc);
  }
}

// ---------------------------------------------------------------------------
// Newton-Schulz(5) on 64x512 matrices; one block per matrix (8 dwv^T + 8 dwo).
// X kept in f16 (row-major + transposed) in ws; A,B tiles in LDS; all matmuls WMMA.
__global__ __launch_bounds__(256) void k_ns(
    const float* rawdwv, const float* rawdwo, _Float16* Xbuf, _Float16* XTbuf,
    float* dwv_o, float* dwo_o) {
  int mat = blockIdx.x;
  bool isV = mat < 8;
  int h = isV ? mat : mat - 8;
  const float* G = isV ? (rawdwv + (size_t)h * D * DH) : (rawdwo + (size_t)h * DH * D);
  _Float16* X  = Xbuf + (size_t)mat * DH * D;   // [64][512]
  _Float16* XT = XTbuf + (size_t)mat * D * DH;  // [512][64]
  __shared__ float red[256];
  __shared__ __align__(16) _Float16 Ah[64 * 64];
  __shared__ __align__(16) _Float16 Bh[64 * 64];
  int tid = (int)threadIdx.x, w = tid >> 5, l = tid & 31;
  float s = 0.f;
  for (int idx = tid; idx < DH * D; idx += 256) { float v = G[idx]; s += v * v; }
  red[tid] = s; __syncthreads();
  for (int off = 128; off; off >>= 1) {
    if (tid < off) red[tid] += red[tid + off];
    __syncthreads();
  }
  float inv = 1.f / fmaxf(sqrtf(red[0]), 1e-7f);
  for (int idx = tid; idx < DH * D; idx += 256) {
    int e = idx / D, d = idx % D;
    float v = (isV ? G[(size_t)d * DH + e] : G[idx]) * inv;  // dwv is transposed into X
    X[idx] = (_Float16)v;
    XT[(size_t)d * DH + e] = (_Float16)v;
  }
  __syncthreads();
  const float CA = 3.4445f, CB = -4.775f, CC = 2.0315f;
  for (int step = 0; step < 5; ++step) {
    // A = X X^T (Bt of X^T is X itself)
    v8f Af[2];
#pragma unroll
    for (int t = 0; t < 2; ++t) {
      int id = w * 2 + t, m0 = (id >> 2) * 16, n0 = (id & 3) * 16;
      v8f a = vzero();
      for (int k = 0; k < D; k += 32)
        a = wmma16(frag_a(X, D, m0, k), frag_b(X, D, n0, k), a);
      Af[t] = a;
      int n = n0 + (l & 15), mbb = m0 + ((l >> 4) << 3);
#pragma unroll
      for (int i = 0; i < 8; ++i) Ah[(mbb + i) * 64 + n] = (_Float16)a[i];
    }
    __syncthreads();
    // B = CB*A + CC*A@A  (A symmetric => Bt of A is A)
#pragma unroll
    for (int t = 0; t < 2; ++t) {
      int id = w * 2 + t, m0 = (id >> 2) * 16, n0 = (id & 3) * 16;
      v8f a2 = vzero();
      for (int k = 0; k < 64; k += 32)
        a2 = wmma16(frag_a(Ah, 64, m0, k), frag_b(Ah, 64, n0, k), a2);
      int n = n0 + (l & 15), mbb = m0 + ((l >> 4) << 3);
#pragma unroll
      for (int i = 0; i < 8; ++i)
        Bh[(mbb + i) * 64 + n] = (_Float16)(CB * Af[t][i] + CC * a2[i]);
    }
    __syncthreads();
    // X = CA*X + B @ X ; each wave owns a disjoint 64-column panel (column-local).
    v8f acc[16];
#pragma unroll
    for (int mi = 0; mi < 4; ++mi)
#pragma unroll
      for (int ni = 0; ni < 4; ++ni) {
        v8f a = vzero();
        int n0 = w * 64 + ni * 16;
        for (int k = 0; k < 64; k += 32)
          a = wmma16(frag_a(Bh, 64, mi * 16, k), frag_b(XT, DH, n0, k), a);
        acc[mi * 4 + ni] = a;
      }
#pragma unroll
    for (int mi = 0; mi < 4; ++mi)
#pragma unroll
      for (int ni = 0; ni < 4; ++ni) {
        int n = w * 64 + ni * 16 + (l & 15);
        int mbb = mi * 16 + ((l >> 4) << 3);
#pragma unroll
        for (int i = 0; i < 8; ++i) {
          float v = CA * (float)X[(size_t)(mbb + i) * D + n] + acc[mi * 4 + ni][i];
          X[(size_t)(mbb + i) * D + n] = (_Float16)v;
          XT[(size_t)n * DH + (mbb + i)] = (_Float16)v;
        }
      }
    __syncthreads();
  }
  for (int idx = tid; idx < DH * D; idx += 256) {
    int e = idx / D, d = idx % D;
    float v = (float)X[idx];
    if (isV) dwv_o[(size_t)h * D * DH + (size_t)d * DH + e] = v;   // transpose back
    else     dwo_o[(size_t)h * DH * D + idx] = v;
  }
}

// ---------------------------------------------------------------------------
extern "C" void kernel_launch(void* const* d_in, const int* in_sizes, int n_in,
                              void* d_out, int out_size, void* d_ws, size_t ws_size,
                              hipStream_t stream) {
  (void)in_sizes; (void)n_in;
  const float* tokens = (const float*)d_in[0];
  const float* rmsw   = (const float*)d_in[1];
  const float* wq     = (const float*)d_in[2];
  const float* wk     = (const float*)d_in[3];
  const float* wv     = (const float*)d_in[4];
  const float* wo     = (const float*)d_in[5];
  const float* lrw    = (const float*)d_in[6];
  const float* tvw    = (const float*)d_in[7];

  char* ws = (char*)d_ws;
  size_t off = 0;
  auto alloc = [&](size_t bytes) -> void* {
    void* p = ws + off;
    off += (bytes + 255) & ~(size_t)255;
    return p;
  };
  const size_t BND = (size_t)B * N * D;        // 4096*512
  const size_t BHNE = (size_t)B * H * N * DH;  // 2*8*2048*64
  float*    t32     = (float*)alloc(BND * 4);
  _Float16* t16     = (_Float16*)alloc(BND * 2);
  _Float16* tT16    = (_Float16*)alloc(BND * 2);
  _Float16* q16     = (_Float16*)alloc(BHNE * 2);
  _Float16* k16     = (_Float16*)alloc(BHNE * 2);
  _Float16* v16     = (_Float16*)alloc(BHNE * 2);
  _Float16* qT16    = (_Float16*)alloc(BHNE * 2);
  _Float16* kT16    = (_Float16*)alloc(BHNE * 2);
  _Float16* vT16    = (_Float16*)alloc(BHNE * 2);
  _Float16* out16   = (_Float16*)alloc(BND * 2);
  _Float16* outT16  = (_Float16*)alloc(BHNE * 2);
  float*    lse     = (float*)alloc((size_t)B * H * N * 4);
  float*    delta   = (float*)alloc((size_t)B * H * N * 4);
  float*    tv      = (float*)alloc(BND * 4);
  _Float16* err16   = (_Float16*)alloc(BND * 2);
  _Float16* errT16  = (_Float16*)alloc(BND * 2);
  _Float16* dout16  = (_Float16*)alloc(BHNE * 2);
  _Float16* doutT16 = (_Float16*)alloc(BHNE * 2);
  _Float16* dqT16   = (_Float16*)alloc(BHNE * 2);
  _Float16* dkT16   = (_Float16*)alloc(BHNE * 2);
  _Float16* dvT16   = (_Float16*)alloc(BHNE * 2);
  float*    rawdwv  = (float*)alloc((size_t)H * D * DH * 4);
  float*    rawdwo  = (float*)alloc((size_t)H * DH * D * 4);
  _Float16* Xbuf    = (_Float16*)alloc((size_t)16 * DH * D * 2);
  _Float16* XTbuf   = (_Float16*)alloc((size_t)16 * D * DH * 2);
  _Float16* wqT16   = (_Float16*)alloc((size_t)H * DH * D * 2);
  _Float16* wkT16   = (_Float16*)alloc((size_t)H * DH * D * 2);
  _Float16* wvT16   = (_Float16*)alloc((size_t)H * DH * D * 2);
  _Float16* wo16    = (_Float16*)alloc((size_t)D * D * 2);
  _Float16* woT16   = (_Float16*)alloc((size_t)D * D * 2);
  _Float16* tvw16   = (_Float16*)alloc((size_t)D * D * 2);
  if (off > ws_size) return;  // workspace too small: do nothing (deterministic)

  float* pred  = (float*)d_out;
  float* dwq_o = pred + BND;
  float* dwk_o = dwq_o + (size_t)H * D * DH;
  float* dwv_o = dwk_o + (size_t)H * D * DH;
  float* dwo_o = dwv_o + (size_t)H * D * DH;
  bool full = (size_t)out_size > BND;

  k_prep<<<5120, 256, 0, stream>>>(wq, wk, wv, wo, tvw, wqT16, wkT16, wvT16, wo16, woT16, tvw16);
  k_rmsnorm<<<B * N, 256, 0, stream>>>(tokens, rmsw, t32, t16, tT16);
  k_qkv<<<dim3(N / 64, H, B), 256, 0, stream>>>(t16, wqT16, wkT16, wvT16,
                                                q16, k16, v16, qT16, kT16, vT16);
  k_attn_fwd<<<dim3(N / 64, H, B), 128, 0, stream>>>(q16, k16, vT16, out16, outT16, lse);
  k_gemm_pred<<<1024, 256, 0, stream>>>(out16, woT16, pred);
  if (!full) return;

  k_gemm_tv<<<1024, 256, 0, stream>>>(t16, tvw16, tv);
  k_error<<<B * N, 256, 0, stream>>>(tv, pred, t32, lrw, err16, errT16);
  k_gemm_dout<<<1024, 256, 0, stream>>>(err16, wo16, dout16, doutT16);
  k_delta<<<4096, 256, 0, stream>>>(dout16, out16, delta);
  k_bwd_dq<<<dim3(N / 64, H, B), 128, 0, stream>>>(q16, k16, v16, kT16, dout16,
                                                   lse, delta, dqT16);
  k_bwd_dkv<<<dim3(N / 64, H, B), 128, 0, stream>>>(q16, k16, v16, qT16, dout16, doutT16,
                                                    lse, delta, dkT16, dvT16);
  k_dw<<<512, 256, 0, stream>>>(tT16, dqT16, dkT16, dvT16, outT16, errT16,
                                dwq_o, dwk_o, rawdwv, rawdwo);
  k_ns<<<16, 256, 0, stream>>>(rawdwv, rawdwo, Xbuf, XTbuf, dwv_o, dwo_o);
}
